// HyperbubbleGNN_56521769615366
// MI455X (gfx1250) — compile-verified, matrix operand order checked by reading, not run on hardware
//
#include <hip/hip_runtime.h>

typedef __attribute__((ext_vector_type(16))) _Float16 v16h;
typedef __attribute__((ext_vector_type(8)))  float    v8f;
typedef __attribute__((ext_vector_type(4)))  unsigned int u32x4;
typedef __attribute__((ext_vector_type(8)))  int      i32x8;
typedef __attribute__((ext_vector_type(4)))  int      i32x4;

union Frag32 { v16h h; uint4 u[2]; };

#define WMMA(a,b,c) __builtin_amdgcn_wmma_f32_16x16x32_f16(false,(a),false,(b),(short)0,(c),false,false)

// A-fragment: lane l<16 -> row l, K = kb*32 + {0..7,16..23}; lane l+16 -> K = kb*32 + {8..15,24..31}.
// Row is 64 halfs (128B); each chunk is 16B-aligned -> two uint4 loads.
__device__ __forceinline__ v16h a_frag(const _Float16* row, int kb, int lane) {
  int base = (kb << 5) + ((lane & 16) ? 8 : 0);
  Frag32 f;
  f.u[0] = *(const uint4*)(row + base);
  f.u[1] = *(const uint4*)(row + base + 16);
  return f.h;
}

// B-fragments are pre-swizzled: frag f occupies 512 halfs; lane's 16 halfs contiguous.
__device__ __forceinline__ v16h b_frag_g(const _Float16* swz, int fid, int lane) {
  const uint4* p = (const uint4*)swz + fid * 64 + lane * 2;
  Frag32 f; f.u[0] = p[0]; f.u[1] = p[1]; return f.h;
}

// ---------------- weight swizzle: f32 [K,64] -> f16 fragments (K padded w/ zeros) ----
// dst[f*512 + lane*16 + j] = W[kb*32 + klocal(lane,j), nt*16 + lane%16]
__global__ void swizzle_weights(const float* __restrict__ w1, const float* __restrict__ w2,
                                const float* __restrict__ m1,
                                _Float16* __restrict__ s1, _Float16* __restrict__ s2,
                                _Float16* __restrict__ sm) {
  int idx = blockIdx.x * blockDim.x + threadIdx.x;
  const float* src; _Float16* dst; int Krows; int local;
  if (idx < 4096)        { src = w1; dst = s1; Krows = 33;  local = idx; }
  else if (idx < 8192)   { src = w2; dst = s2; Krows = 64;  local = idx - 4096; }
  else if (idx < 18432)  { src = m1; dst = sm; Krows = 133; local = idx - 8192; }
  else return;
  int f = local >> 9, r = local & 511, lane = r >> 4, j = r & 15;
  int kb = f >> 2, nt = f & 3;
  int klocal = ((j >> 3) << 4) + (j & 7) + ((lane & 16) ? 8 : 0);
  int k = kb * 32 + klocal;
  int n = nt * 16 + (lane & 15);
  float v = (k < Krows) ? src[k * 64 + n] : 0.f;
  dst[local] = (_Float16)v;
}

// ---------------- node encoder: one-hot conv == per-position table add ---------------
// lane == output channel (32 channels == 32 lanes). wt[t*3+p] = cnn_w[o,t,p].
__global__ void node_encoder(const int* __restrict__ tok, const float* __restrict__ xcov,
                             const float* __restrict__ cw, const float* __restrict__ cb,
                             _Float16* __restrict__ X0h, int N) {
  __shared__ int toks[8][128];
  int lane = threadIdx.x & 31;
  int w    = threadIdx.x >> 5;
  int gw   = (blockIdx.x * blockDim.x + threadIdx.x) >> 5;
  int nw   = (gridDim.x * blockDim.x) >> 5;
  float wt[18];
#pragma unroll
  for (int i = 0; i < 18; i++) wt[i] = cw[lane * 18 + i];
  float bias = cb[lane];
  for (int node = gw; node < N; node += nw) {
    const int* tp = tok + (size_t)node * 128;
#pragma unroll
    for (int i = 0; i < 4; i++) toks[w][lane + 32 * i] = tp[lane + 32 * i];
    __builtin_amdgcn_wave_barrier();  // DS ops are in-order within a wave
    float acc = 0.f;
    int tprev = 0, tcur = toks[w][0];
    for (int k = 0; k < 128; k++) {
      int tnext = (k < 127) ? toks[w][k + 1] : 0;
      float h = wt[tcur * 3 + 1];
      if (k > 0)   h += wt[tprev * 3 + 0];
      if (k < 127) h += wt[tnext * 3 + 2];
      acc += fmaxf(h + bias, 0.f);
      tprev = tcur; tcur = tnext;
    }
    _Float16* orow = X0h + (size_t)node * 64;
    orow[lane]      = (_Float16)(acc * (1.0f / 128.0f));
    orow[32 + lane] = (lane == 0) ? (_Float16)xcov[node] : (_Float16)0.f;  // col32=cov, 33..63=pad
    __builtin_amdgcn_wave_barrier();
  }
}

// ---------------- degree / rsqrt -----------------------------------------------------
__global__ void deg_init(float* deg, int N) {
  int i = blockIdx.x * blockDim.x + threadIdx.x;
  if (i < N) deg[i] = 1.f;  // self loop
}
__global__ void deg_accum(const int* __restrict__ dst, float* deg, int E) {
  int e = blockIdx.x * blockDim.x + threadIdx.x;
  if (e < E) atomicAdd(&deg[dst[e]], 1.f);
}
__global__ void deg_finalize(float* deg, int N) {
  int i = blockIdx.x * blockDim.x + threadIdx.x;
  if (i < N) { float d = deg[i]; deg[i] = d > 0.f ? rsqrtf(d) : 0.f; }
}

// ---------------- [N,64] = [N,64pad] x [64,64] via WMMA, one wave per 16-row tile -----
__global__ void gemm_nx64(const _Float16* __restrict__ A, const _Float16* __restrict__ Bswz,
                          float* __restrict__ C, int ntiles) {
  int lane = threadIdx.x & 31;
  int gw = (blockIdx.x * blockDim.x + threadIdx.x) >> 5;
  int nw = (gridDim.x * blockDim.x) >> 5;
  v16h bf[2][4];
#pragma unroll
  for (int kb = 0; kb < 2; kb++)
#pragma unroll
    for (int nt = 0; nt < 4; nt++) bf[kb][nt] = b_frag_g(Bswz, kb * 4 + nt, lane);
  for (int tile = gw; tile < ntiles; tile += nw) {
    const _Float16* arow = A + (size_t)(tile * 16 + (lane & 15)) * 64;
    v16h a0 = a_frag(arow, 0, lane);
    v16h a1 = a_frag(arow, 1, lane);
    int rowbase = tile * 16 + ((lane & 16) ? 8 : 0);
#pragma unroll
    for (int nt = 0; nt < 4; nt++) {
      v8f c = {};
      c = WMMA(a0, bf[0][nt], c);
      c = WMMA(a1, bf[1][nt], c);
      int col = nt * 16 + (lane & 15);
#pragma unroll
      for (int r = 0; r < 8; r++) C[(size_t)(rowbase + r) * 64 + col] = c[r];
    }
  }
}

// ---------------- aggregation: self-loop init, then edge scatter (f32 atomics) --------
__global__ void agg_self(const float* __restrict__ XW, const float* __restrict__ dinv,
                         float* __restrict__ OUT, int N) {
  int i = blockIdx.x * blockDim.x + threadIdx.x;
  if (i < N * 64) { float d = dinv[i >> 6]; OUT[i] = XW[i] * d * d; }
}
__global__ void agg_edges(const int* __restrict__ esrc, const int* __restrict__ edst,
                          const float* __restrict__ dinv, const float* __restrict__ H,
                          float* __restrict__ OUT, int E) {
  int lane = threadIdx.x & 31;
  int gw = (blockIdx.x * blockDim.x + threadIdx.x) >> 5;
  int nw = (gridDim.x * blockDim.x) >> 5;
  for (int e = gw; e < E; e += nw) {
    int s = esrc[e], d = edst[e];
    float norm = dinv[s] * dinv[d];
    const float* hs = H + (size_t)s * 64;
    float* od = OUT + (size_t)d * 64;
    atomicAdd(&od[lane],      hs[lane]      * norm);
    atomicAdd(&od[lane + 32], hs[lane + 32] * norm);
  }
}
__global__ void bias_relu_h(const float* __restrict__ AGG, const float* __restrict__ bias,
                            _Float16* __restrict__ Hh, int N) {
  int i = blockIdx.x * blockDim.x + threadIdx.x;
  if (i < N * 64) Hh[i] = (_Float16)fmaxf(AGG[i] + bias[i & 63], 0.f);
}

// ---------------- edge MLP: [16 edges x 160] x [160 x 64] WMMA + fused 64->1 layer ----
// Weight fragments (20KB) are staged into LDS by the Tensor Data Mover: one
// tensor_load_to_lds issued by wave 0 (TDM ignores EXEC; wave-uniform guard
// gives exactly one issue per workgroup), fenced by s_wait_tensorcnt + barrier.
__global__ void edge_mlp(const int* __restrict__ esrc, const int* __restrict__ edst,
                         const float* __restrict__ eattr, const _Float16* __restrict__ Hh,
                         const _Float16* __restrict__ M1swz, const float* __restrict__ b1v,
                         const float* __restrict__ w2v, const float* __restrict__ b2v,
                         float* __restrict__ out, int etiles) {
  __shared__ uint4 lb[1280];  // 20 fragments * 512 halfs = 20480 bytes
  if (threadIdx.x < 32) {
    // Generic pointers into the LDS aperture carry the raw LDS offset in the
    // low 32 bits (ISA 10.2 aperture mapping).
    unsigned lds_addr = (unsigned)(uintptr_t)&lb[0];
    unsigned long long ga = (unsigned long long)(uintptr_t)M1swz;
    // D# group 0: count=1 (valid), lds_addr, 57-bit global addr, type=2.
    u32x4 g0 = { 1u,
                 lds_addr,
                 (unsigned)(ga & 0xffffffffu),
                 (unsigned)((ga >> 32) & 0x01ffffffu) | (2u << 30) };
    // D# group 1: data_size=2 (4B); tensor_dim0=5120, tensor_dim1=1;
    // tile_dim0=5120, tile_dim1=1; tensor_dim0_stride=5120 -> 20480B copy.
    i32x8 g1 = { (int)0x00020000u,                  // wg_mask=0, data_size=4B
                 (int)((5120u & 0xffffu) << 16),    // tensor_dim0[15:0] @ 63:48
                 (int)(1u << 16),                   // tensor_dim0 hi=0, tensor_dim1 lo=1
                 (int)(5120u << 16),                // tensor_dim1 hi=0, tile_dim0=5120
                 1,                                 // tile_dim1=1, tile_dim2=0
                 5120,                              // tensor_dim0_stride[31:0]
                 0, 0 };
    i32x4 z4 = { 0, 0, 0, 0 };
    i32x8 z8 = { 0, 0, 0, 0, 0, 0, 0, 0 };
    // 6-arg form (this toolchain): (g0, g1, g2, g3, g4, cpol)
    __builtin_amdgcn_tensor_load_to_lds(g0, g1, z4, z4, z8, 0);
    __builtin_amdgcn_s_wait_tensorcnt(0);
  }
  __syncthreads();
  int lane = threadIdx.x & 31;
  int gw = (blockIdx.x * blockDim.x + threadIdx.x) >> 5;
  int nw = (gridDim.x * blockDim.x) >> 5;
  float b2 = b2v[0];
  for (int tile = gw; tile < etiles; tile += nw) {
    int e = tile * 16 + (lane & 15);   // lane l and l+16 own the same edge row
    int u = esrc[e], v = edst[e];
    v16h a0 = a_frag(Hh + (size_t)u * 64, 0, lane);
    v16h a1 = a_frag(Hh + (size_t)u * 64, 1, lane);
    v16h a2 = a_frag(Hh + (size_t)v * 64, 0, lane);
    v16h a3 = a_frag(Hh + (size_t)v * 64, 1, lane);
    Frag32 fe; fe.u[0] = make_uint4(0, 0, 0, 0); fe.u[1] = make_uint4(0, 0, 0, 0);
    if (lane < 16) {  // K-block 4: klocal 0..4 = edge_attr, rest zero-pad
      const float* ap = eattr + (size_t)e * 5;
      fe.h[0] = (_Float16)ap[0]; fe.h[1] = (_Float16)ap[1]; fe.h[2] = (_Float16)ap[2];
      fe.h[3] = (_Float16)ap[3]; fe.h[4] = (_Float16)ap[4];
    }
    v16h a4 = fe.h;
    float part[8];
#pragma unroll
    for (int r = 0; r < 8; r++) part[r] = 0.f;
#pragma unroll
    for (int nt = 0; nt < 4; nt++) {
      v8f c = {};
#pragma unroll
      for (int kb = 0; kb < 5; kb++) {
        Frag32 fb;
        int fid = kb * 4 + nt;
        fb.u[0] = lb[fid * 64 + lane * 2];
        fb.u[1] = lb[fid * 64 + lane * 2 + 1];
        v16h a = (kb == 0) ? a0 : (kb == 1) ? a1 : (kb == 2) ? a2 : (kb == 3) ? a3 : a4;
        c = WMMA(a, fb.h, c);
      }
      int n = nt * 16 + (lane & 15);
      float b1 = b1v[n], w2 = w2v[n];
#pragma unroll
      for (int r = 0; r < 8; r++) part[r] += fmaxf(c[r] + b1, 0.f) * w2;
    }
#pragma unroll
    for (int m = 1; m < 16; m <<= 1)
#pragma unroll
      for (int r = 0; r < 8; r++) part[r] += __shfl_xor(part[r], m, 32);
    if ((lane & 15) == 0) {  // lane0 -> rows 0..7, lane16 -> rows 8..15
      float* op = out + tile * 16 + ((lane & 16) ? 8 : 0);
#pragma unroll
      for (int r = 0; r < 8; r++) op[r] = part[r] + b2;
    }
  }
}

extern "C" void kernel_launch(void* const* d_in, const int* in_sizes, int n_in,
                              void* d_out, int out_size, void* d_ws, size_t ws_size,
                              hipStream_t stream) {
  const float* xcov  = (const float*)d_in[1];
  const int*   eidx  = (const int*)  d_in[2];
  const float* eattr = (const float*)d_in[3];
  const float* cnnw  = (const float*)d_in[4];
  const float* cnnb  = (const float*)d_in[5];
  const float* g1w   = (const float*)d_in[6];
  const float* g1b   = (const float*)d_in[7];
  const float* g2w   = (const float*)d_in[8];
  const float* g2b   = (const float*)d_in[9];
  const float* m1w   = (const float*)d_in[10];
  const float* m1b   = (const float*)d_in[11];
  const float* m2w   = (const float*)d_in[12];
  const float* m2b   = (const float*)d_in[13];
  float* out = (float*)d_out;

  const int N = in_sizes[1];       // x_cov is [N,1]
  const int E = in_sizes[2] / 2;   // edge_index is [2,E]
  const int* esrc = eidx;
  const int* edst = eidx + E;

  char* ws = (char*)d_ws; size_t off = 0;
  auto take = [&](size_t bytes) -> void* {
    void* p = ws + off;
    off += (bytes + 255) & ~(size_t)255;
    return p;
  };
  _Float16* X0h = (_Float16*)take((size_t)N * 64 * 2);
  _Float16* H1h = (_Float16*)take((size_t)N * 64 * 2);
  _Float16* H2h = (_Float16*)take((size_t)N * 64 * 2);
  float*    XW  = (float*)   take((size_t)N * 64 * 4);
  float*    AGG = (float*)   take((size_t)N * 64 * 4);
  float*    dnv = (float*)   take((size_t)N * 4);   // deg, then in-place rsqrt
  _Float16* W1s = (_Float16*)take(8  * 512 * 2);
  _Float16* W2s = (_Float16*)take(8  * 512 * 2);
  _Float16* M1s = (_Float16*)take(20 * 512 * 2);

  int ntiles = N / 16;   // N = 100000 = 6250*16
  int etiles = E / 16;   // E = 1600000 = 100000*16
  int nEl = N * 64;

  swizzle_weights<<<72, 256, 0, stream>>>(g1w, g2w, m1w, W1s, W2s, M1s);
  node_encoder<<<4096, 256, 0, stream>>>((const int*)d_in[0], xcov, cnnw, cnnb, X0h, N);
  deg_init    <<<(N + 255) / 256, 256, 0, stream>>>(dnv, N);
  deg_accum   <<<(E + 255) / 256, 256, 0, stream>>>(edst, dnv, E);
  deg_finalize<<<(N + 255) / 256, 256, 0, stream>>>(dnv, N);

  // ---- GCN layer 1 ----
  gemm_nx64<<<(ntiles + 7) / 8, 256, 0, stream>>>(X0h, W1s, XW, ntiles);
  agg_self <<<(nEl + 255) / 256, 256, 0, stream>>>(XW, dnv, AGG, N);
  agg_edges<<<16384, 256, 0, stream>>>(esrc, edst, dnv, XW, AGG, E);
  bias_relu_h<<<(nEl + 255) / 256, 256, 0, stream>>>(AGG, g1b, H1h, N);

  // ---- GCN layer 2 ----
  gemm_nx64<<<(ntiles + 7) / 8, 256, 0, stream>>>(H1h, W2s, XW, ntiles);
  agg_self <<<(nEl + 255) / 256, 256, 0, stream>>>(XW, dnv, AGG, N);
  agg_edges<<<16384, 256, 0, stream>>>(esrc, edst, dnv, XW, AGG, E);
  bias_relu_h<<<(nEl + 255) / 256, 256, 0, stream>>>(AGG, g2b, H2h, N);

  // ---- edge classifier (dominant FLOPs, WMMA + TDM-staged weights) ----
  edge_mlp<<<4096, 256, 0, stream>>>(esrc, edst, eattr, H2h, M1s, m1b, m2w, m2b, out, etiles);
}
